// GraphBackboneMPNN_40956808135088
// MI455X (gfx1250) — compile-verified, matrix operand order checked by reading, not run on hardware
//
#include <hip/hip_runtime.h>

typedef __attribute__((ext_vector_type(16))) __bf16 v16bf;
typedef __attribute__((ext_vector_type(8)))  __bf16 v8bf;
typedef __attribute__((ext_vector_type(8)))  float  v8f;

#define N_NODES 4096
#define N_EDGES 16384
#define N_GRAPH 128
#define HDIM    64
#define EDIM    16
#define HD      4096      // edge-MLP hidden = ic*oc = 64*64
#define EB      64        // edges per block
#define KC      64        // k-chunk staged in LDS (keeps VGPRs < 256)
#define H1PAD   (KC + 8)  // 72 bf16 -> 144B row stride, conflict-free b128 reads

__device__ __forceinline__ __bf16 f2bf(float f) { return (__bf16)f; }

__device__ __forceinline__ v16bf ld_a16(const __bf16* p) {
  v8bf lo = *(const v8bf*)p;
  v8bf hh = *(const v8bf*)(p + 8);
  return __builtin_shufflevector(lo, hh, 0,1,2,3,4,5,6,7,8,9,10,11,12,13,14,15);
}

__global__ __launch_bounds__(256) void k_zero(float* __restrict__ p, int n) {
  int i = blockIdx.x * 256 + threadIdx.x;
  if (i < n) p[i] = 0.f;
}

__global__ __launch_bounds__(256) void k_count(const int* __restrict__ idx,
                                               float* __restrict__ cnt, int n) {
  int i = blockIdx.x * 256 + threadIdx.x;
  if (i < n) unsafeAtomicAdd(&cnt[idx[i]], 1.0f);
}

// w2 [k=4096][io=4096] fp32 -> w2t [io][k] bf16 (tiled transpose via LDS)
__global__ __launch_bounds__(256) void k_w2t(const float* __restrict__ w2,
                                             __bf16* __restrict__ w2t) {
  __shared__ float tile[64][65];
  const int bk = blockIdx.x & 63;
  const int bi = blockIdx.x >> 6;
  const int tid = threadIdx.x;
  for (int t = tid; t < 4096; t += 256) {
    int kk = t >> 6, ii = t & 63;
    tile[kk][ii] = w2[(size_t)(bk * 64 + kk) * HD + bi * 64 + ii];
  }
  __syncthreads();
  for (int t = tid; t < 4096; t += 256) {
    int ii = t >> 6, kk = t & 63;
    w2t[(size_t)(bi * 64 + ii) * HD + bk * 64 + kk] = f2bf(tile[kk][ii]);
  }
}

// Fused per-edge kernel: msg[e,:] = x[src[e],:] @ reshape(relu(ea@w1+b1)@w2 + b2, 64,64)
// scattered (sum) into agg[dst[e],:] with f32 atomics.
// Block = 64 edges, 8 waves. wave -> (ntile = wave&3 : 16 oc cols, ihalf = wave>>2 : 32 i's).
// Each wave computes all 4 edge-row tiles: every B tile is loaded exactly once per block,
// and B is double-buffered in registers so WMMAs never wait on loadcnt==0.
__global__ __launch_bounds__(256) void k_edge(
    const float* __restrict__ xin, const int* __restrict__ src, const int* __restrict__ dst,
    const float* __restrict__ ea,  const float* __restrict__ w1, const float* __restrict__ b1,
    const __bf16* __restrict__ w2t, const float* __restrict__ b2, float* __restrict__ agg)
{
  __shared__ float  s_x[EB][HDIM];     // gathered source features  (16 KB)
  __shared__ float  s_ea[EB][EDIM];    // edge attrs                ( 4 KB)
  __shared__ float  s_w1[EDIM][KC];    // w1 chunk                  ( 4 KB)
  __shared__ float  s_b1[KC];
  __shared__ __bf16 s_h1[EB][H1PAD];   // relu(ea@w1+b1) chunk      ( 9 KB)
  __shared__ int    s_dst[EB];

  const int tid  = threadIdx.x;
  const int eb0  = blockIdx.x * EB;
  const int lane = tid & 31;
  const int wave = tid >> 5;
  const int ln   = lane & 15;
  const int hi   = lane >> 4;           // k/row half select per WMMA layout
  const int ntile = wave & 3;           // output-column tile (16 oc)
  const int ibase = (wave >> 2) * 32;   // this wave's i-range [ibase, ibase+32)
  const int col  = ntile * 16 + ln;     // oc column 0..63

  if (tid < EB) s_dst[tid] = dst[eb0 + tid];
  for (int t = tid; t < EB * EDIM; t += 256)
    s_ea[t >> 4][t & 15] = ea[(size_t)eb0 * EDIM + t];
  for (int t = tid; t < EB * HDIM; t += 256) {
    int e = t >> 6, c = t & 63;
    s_x[e][c] = xin[(size_t)src[eb0 + e] * HDIM + c];
  }

  v8f acc[4] = {};                      // msg accumulators, one per m-tile

  const int ce = tid >> 2;              // h1-compute mapping: edge
  const int q  = tid & 3;               // k slot (keeps LDS stores conflict-free)

  for (int kc = 0; kc < HD; kc += KC) {
    __syncthreads();                    // prior chunk's WMMA reads of s_h1 done
    for (int t = tid; t < EDIM * KC; t += 256)
      s_w1[t >> 6][t & 63] = w1[(size_t)(t >> 6) * HD + kc + (t & 63)];
    if (tid < KC) s_b1[tid] = b1[kc + tid];
    __syncthreads();

    // h1 chunk = relu(ea @ w1 + b1), stored bf16 in LDS (WMMA-A friendly rows)
    float eav[EDIM];
    #pragma unroll
    for (int d = 0; d < EDIM; ++d) eav[d] = s_ea[ce][d];
    #pragma unroll 4
    for (int j = 0; j < 8; ++j) {
      int k0 = 2 * (q + 4 * j);
      float s0 = s_b1[k0], s1 = s_b1[k0 + 1];
      #pragma unroll
      for (int d = 0; d < EDIM; ++d) {
        s0 += eav[d] * s_w1[d][k0];
        s1 += eav[d] * s_w1[d][k0 + 1];
      }
      s0 = s0 > 0.f ? s0 : 0.f;
      s1 = s1 > 0.f ? s1 : 0.f;
      unsigned u = (unsigned)__builtin_bit_cast(unsigned short, f2bf(s0))
                 | ((unsigned)__builtin_bit_cast(unsigned short, f2bf(s1)) << 16);
      *(unsigned*)&s_h1[ce][k0] = u;
    }
    __syncthreads();

    // Hoist all A tiles for this chunk into registers (4 m-tiles x 2 ksteps = 64 VGPRs).
    v16bf A[4][2];
    #pragma unroll
    for (int m = 0; m < 4; ++m) {
      const __bf16* ap = &s_h1[m * 16 + ln][hi * 16];
      #pragma unroll
      for (int ks = 0; ks < 2; ++ks) A[m][ks] = ld_a16(ap + ks * 32);
    }

    const __bf16* bbase = w2t + (size_t)(ibase * 64 + col) * HD + kc + hi * 16;
    v16bf B0[2], B1[2];
    auto loadB = [&](int i, v16bf* B) {
      const __bf16* bp = bbase + (size_t)i * (64 * (size_t)HD);
      #pragma unroll
      for (int ks = 0; ks < 2; ++ks) B[ks] = *(const v16bf*)(bp + ks * 32);
    };
    auto compute = [&](int i, const v16bf* B) {
      v8f t0 = {}, t1 = {}, t2 = {}, t3 = {};
      #pragma unroll
      for (int ks = 0; ks < 2; ++ks) {
        t0 = __builtin_amdgcn_wmma_f32_16x16x32_bf16(false, A[0][ks], false, B[ks], (short)0, t0, false, false);
        t1 = __builtin_amdgcn_wmma_f32_16x16x32_bf16(false, A[1][ks], false, B[ks], (short)0, t1, false, false);
        t2 = __builtin_amdgcn_wmma_f32_16x16x32_bf16(false, A[2][ks], false, B[ks], (short)0, t2, false, false);
        t3 = __builtin_amdgcn_wmma_f32_16x16x32_bf16(false, A[3][ks], false, B[ks], (short)0, t3, false, false);
      }
      // msg += x[:, i] (row scale) * partial theta tiles; linear in k so chunking is exact
      const int ii = ibase + i;
      #pragma unroll
      for (int r = 0; r < 8; ++r) {
        acc[0][r] = __builtin_fmaf(s_x[ 0 + hi * 8 + r][ii], t0[r], acc[0][r]);
        acc[1][r] = __builtin_fmaf(s_x[16 + hi * 8 + r][ii], t1[r], acc[1][r]);
        acc[2][r] = __builtin_fmaf(s_x[32 + hi * 8 + r][ii], t2[r], acc[2][r]);
        acc[3][r] = __builtin_fmaf(s_x[48 + hi * 8 + r][ii], t3[r], acc[3][r]);
      }
    };

    loadB(0, B0);
    #pragma unroll 1
    for (int i = 0; i < 32; i += 2) {
      loadB(i + 1, B1);                       // prefetch while B0 WMMAs run
      compute(i, B0);
      loadB(i + 2 < 32 ? i + 2 : 0, B0);      // prefetch (tail reload is harmless)
      compute(i + 1, B1);
    }
  }

  // b2 contribution over this wave's i-range: msg += x_src[:, i] * b2[i*64+col]
  #pragma unroll 1
  for (int i = ibase; i < ibase + 32; ++i) {
    float bv = b2[i * 64 + col];
    #pragma unroll
    for (int m = 0; m < 4; ++m) {
      #pragma unroll
      for (int r = 0; r < 8; ++r)
        acc[m][r] = __builtin_fmaf(s_x[m * 16 + hi * 8 + r][i], bv, acc[m][r]);
    }
  }

  // scatter-sum into agg[dst] (both i-half waves add their partial sums)
  #pragma unroll
  for (int m = 0; m < 4; ++m) {
    #pragma unroll
    for (int r = 0; r < 8; ++r) {
      int e = m * 16 + hi * 8 + r;
      unsafeAtomicAdd(&agg[(size_t)s_dst[e] * HDIM + col], acc[m][r]);
    }
  }
}

// xout = relu(agg/max(cnt,1) + xin @ root + bias)
__global__ __launch_bounds__(256) void k_node(
    const float* __restrict__ agg, const float* __restrict__ cnt,
    const float* __restrict__ xin, const float* __restrict__ root,
    const float* __restrict__ bias, float* __restrict__ xout)
{
  __shared__ float s_r[HDIM * HDIM];
  __shared__ float s_b[HDIM];
  const int tid = threadIdx.x;
  for (int t = tid; t < HDIM * HDIM; t += 256) s_r[t] = root[t];
  if (tid < HDIM) s_b[tid] = bias[tid];
  __syncthreads();
  const int node = blockIdx.x * 4 + (tid >> 6);
  const int o = tid & 63;
  const float* xr = xin + (size_t)node * HDIM;
  float s = s_b[o];
  #pragma unroll 8
  for (int i = 0; i < HDIM; ++i) s += xr[i] * s_r[i * HDIM + o];
  float c = cnt[node]; c = c > 1.f ? c : 1.f;
  s += agg[(size_t)node * HDIM + o] / c;
  xout[(size_t)node * HDIM + o] = s > 0.f ? s : 0.f;
}

__global__ __launch_bounds__(256) void k_pool_scatter(
    const float* __restrict__ h, const int* __restrict__ batch, float* __restrict__ out)
{
  int t = blockIdx.x * 256 + threadIdx.x;   // over N*64
  int n = t >> 6, o = t & 63;
  unsafeAtomicAdd(&out[(size_t)batch[n] * HDIM + o], h[t]);
}

__global__ __launch_bounds__(256) void k_pool_div(float* __restrict__ out,
                                                  const float* __restrict__ gcnt) {
  int t = blockIdx.x * 256 + threadIdx.x;   // over G*64
  float c = gcnt[t >> 6]; c = c > 1.f ? c : 1.f;
  out[t] = out[t] / c;
}

extern "C" void kernel_launch(void* const* d_in, const int* in_sizes, int n_in,
                              void* d_out, int out_size, void* d_ws, size_t ws_size,
                              hipStream_t stream)
{
  const float* x   = (const float*)d_in[0];
  const int*   ei  = (const int*)d_in[1];
  const float* ea  = (const float*)d_in[2];
  const int*   bat = (const int*)d_in[3];
  const int* src = ei;
  const int* dst = ei + N_EDGES;

  char* w = (char*)d_ws;
  __bf16* w2t = (__bf16*)w;                                  // 32 MB, reused per layer
  float* agg  = (float*)(w + (size_t)HD * HD * 2);           // 1 MB
  float* cnt  = agg + (size_t)N_NODES * HDIM;                // 16 KB
  float* hA   = cnt + N_NODES;                               // 1 MB
  float* hB   = hA + (size_t)N_NODES * HDIM;                 // 1 MB
  float* gcnt = hB + (size_t)N_NODES * HDIM;                 // 512 B

  k_zero <<<(N_NODES + 255) / 256, 256, 0, stream>>>(cnt, N_NODES);
  k_count<<<N_EDGES / 256, 256, 0, stream>>>(dst, cnt, N_EDGES);

  const float* xin = x;
  float* hbuf[2] = {hA, hB};
  for (int l = 0; l < 3; ++l) {
    const float* w1   = (const float*)d_in[4 + 6 * l + 0];
    const float* b1   = (const float*)d_in[4 + 6 * l + 1];
    const float* w2   = (const float*)d_in[4 + 6 * l + 2];
    const float* b2   = (const float*)d_in[4 + 6 * l + 3];
    const float* root = (const float*)d_in[4 + 6 * l + 4];
    const float* bias = (const float*)d_in[4 + 6 * l + 5];
    k_w2t <<<4096, 256, 0, stream>>>(w2, w2t);
    k_zero<<<(N_NODES * HDIM) / 256, 256, 0, stream>>>(agg, N_NODES * HDIM);
    k_edge<<<N_EDGES / EB, 256, 0, stream>>>(xin, src, dst, ea, w1, b1, w2t, b2, agg);
    k_node<<<N_NODES / 4, 256, 0, stream>>>(agg, cnt, xin, root, bias, hbuf[l & 1]);
    xin = hbuf[l & 1];
  }

  k_zero<<<(N_GRAPH * HDIM) / 256, 256, 0, stream>>>((float*)d_out, N_GRAPH * HDIM);
  k_zero<<<1, 256, 0, stream>>>(gcnt, N_GRAPH);
  k_count<<<N_NODES / 256, 256, 0, stream>>>(bat, gcnt, N_NODES);
  k_pool_scatter<<<(N_NODES * HDIM) / 256, 256, 0, stream>>>(xin, bat, (float*)d_out);
  k_pool_div<<<(N_GRAPH * HDIM) / 256, 256, 0, stream>>>((float*)d_out, gcnt);
}
